// Model_26508538151583
// MI455X (gfx1250) — matrix operand order repaired and checked by reading, not compiled, not run on hardware
//
#include <hip/hip_runtime.h>
#include <hip/hip_bf16.h>

// ---------------------------------------------------------------------------
// GraphSAGE (mean) x2 + edge-score MLP for MI455X (gfx1250, wave32, WMMA).
//   layer:  out = x @ Wself^T + (msg/deg) @ Wneigh^T + b   [+ relu]
//   pred :  relu(cat @ P1^T + b1) -> relu(@ P2^T + b2) -> @ P3^T + b3
// Matrix math in bf16 via V_WMMA_F32_16X16X32_BF16 (f32 accumulate).
// Weights pre-converted to bf16 in workspace once per launch (L2-resident).
// ---------------------------------------------------------------------------

typedef __attribute__((ext_vector_type(16))) __bf16 v16bf;
typedef __attribute__((ext_vector_type(8)))  float  v8f;

#define D_FEAT 128

// Pack 4 f32 -> 4 bf16 (8 bytes) using native conversions (v_cvt_pk_bf16_f32).
__device__ __forceinline__ void store_bf4(__bf16* dst, float4 v) {
    union { __bf16 h[4]; uint2 u; } t;
    t.h[0] = (__bf16)v.x; t.h[1] = (__bf16)v.y;
    t.h[2] = (__bf16)v.z; t.h[3] = (__bf16)v.w;
    *(uint2*)dst = t.u;
}

// A-fragment (16x32 bf16, MxK) from an LDS tile of row stride `stride` bf16.
// ISA layout: lane m=lane&15, half=lane>>4; element e -> k = (e>>3)*16 + half*8 + (e&7)
// -> two contiguous 8-bf16 (16B) ds_load_b128 per fragment.
__device__ __forceinline__ v16bf load_a_frag(const __bf16* lds, int stride,
                                             int m, int half, int kb) {
    union { v16bf v; uint4 q[2]; } u;
    const __bf16* p = lds + m * stride + kb * 32 + half * 8;
    u.q[0] = *(const uint4*)(p);
    u.q[1] = *(const uint4*)(p + 16);
    return u.v;
}

// B-fragment (32x16 bf16, KxN) from bf16 row-major W[row][K]:
// B[k][n] = W[o_base+n][k]; lane n=lane&15, half=lane>>4; element e -> k = kb*32+(e>>3)*16+half*8+(e&7)
// -> two contiguous 8-bf16 (16B) global_load_b128, no conversion.
__device__ __forceinline__ v16bf load_b_frag(const __bf16* __restrict__ Wb, int K,
                                             int row, int half, int kb) {
    union { v16bf v; uint4 q[2]; } u;
    const __bf16* p = Wb + (long long)row * K + kb * 32 + half * 8;
    u.q[0] = *(const uint4*)(p);
    u.q[1] = *(const uint4*)(p + 16);
    return u.v;
}

// ---------------------------------------------------------------------------
// Utility kernels
// ---------------------------------------------------------------------------
__global__ void zero_f32_kernel(float* __restrict__ p, long long n) {
    long long i = (long long)blockIdx.x * blockDim.x + threadIdx.x;
    if (i < n) p[i] = 0.0f;
}

__global__ void f32_to_bf16_kernel(const float* __restrict__ src,
                                   __bf16* __restrict__ dst, int n /*mult of 4*/) {
    int i = (blockIdx.x * blockDim.x + threadIdx.x) * 4;
    if (i < n) store_bf4(dst + i, *(const float4*)(src + i));
}

__global__ void deg_kernel(const int* __restrict__ dst, float* __restrict__ deg, int nE) {
    int e = blockIdx.x * blockDim.x + threadIdx.x;
    if (e < nE) atomicAdd(deg + dst[e], 1.0f);
}

// One 32-lane wave per edge; each lane moves 4 floats (128 = 32*4).
__global__ void scatter_kernel(const float* __restrict__ feat,
                               const int* __restrict__ src, const int* __restrict__ dst,
                               float* __restrict__ msg, int nE) {
    long long tid = (long long)blockIdx.x * blockDim.x + threadIdx.x;
    int e = (int)(tid >> 5);
    if (e >= nE) return;
    int lane = (int)(tid & 31);
    int s = src[e], d = dst[e];
    float4 v = *(const float4*)(feat + (long long)s * D_FEAT + lane * 4);
    float* o = msg + (long long)d * D_FEAT + lane * 4;
    atomicAdd(o + 0, v.x);
    atomicAdd(o + 1, v.y);
    atomicAdd(o + 2, v.z);
    atomicAdd(o + 3, v.w);
}

// ---------------------------------------------------------------------------
// SAGE layer: 16-node tile per block, 4 waves, each wave 2 of 8 column tiles.
// out[n,o] = sum_k self[n,k]*Wself[o,k] + sum_k neigh[n,k]*Wneigh[o,k] + b[o]
// ---------------------------------------------------------------------------
__global__ __launch_bounds__(128)
void sage_layer_kernel(const float* __restrict__ feat, const float* __restrict__ msg,
                       const float* __restrict__ deg,
                       const __bf16* __restrict__ Wself, const __bf16* __restrict__ Wneigh,
                       const float* __restrict__ bias, float* __restrict__ out, int doRelu)
{
    __shared__ __align__(16) __bf16 a_self[16 * D_FEAT];
    __shared__ __align__(16) __bf16 a_neigh[16 * D_FEAT];
    __shared__ float rdeg[16];
    const int tid = threadIdx.x;
    const int node_base = blockIdx.x * 16;

    if (tid < 16) rdeg[tid] = 1.0f / fmaxf(deg[node_base + tid], 1.0f);
    __syncthreads();

    // Stage activations as bf16: 16 rows x 128 ch, 4 elements per iteration.
    for (int i = tid; i < 16 * 32; i += 128) {
        int r = i >> 5, q = (i & 31) * 4;
        long long g = (long long)(node_base + r) * D_FEAT + q;
        float4 xs = *(const float4*)(feat + g);
        float4 ms = *(const float4*)(msg + g);
        float rd = rdeg[r];
        store_bf4(&a_self[r * D_FEAT + q], xs);
        store_bf4(&a_neigh[r * D_FEAT + q],
                  make_float4(ms.x * rd, ms.y * rd, ms.z * rd, ms.w * rd));
    }
    __syncthreads();

    const int lane = tid & 31, wave = tid >> 5;
    const int m = lane & 15, half = lane >> 4;

    #pragma unroll
    for (int t = 0; t < 2; ++t) {
        const int o_base = (wave + t * 4) * 16;
        v8f c = {};
        #pragma unroll
        for (int kb = 0; kb < 4; ++kb) {
            v16bf as = load_a_frag(a_self, D_FEAT, m, half, kb);
            v16bf bs = load_b_frag(Wself, D_FEAT, o_base + m, half, kb);
            c = __builtin_amdgcn_wmma_f32_16x16x32_bf16(false, as, false, bs,
                                                        (short)0, c, false, false);
            v16bf an = load_a_frag(a_neigh, D_FEAT, m, half, kb);
            v16bf bn = load_b_frag(Wneigh, D_FEAT, o_base + m, half, kb);
            c = __builtin_amdgcn_wmma_f32_16x16x32_bf16(false, an, false, bn,
                                                        (short)0, c, false, false);
        }
        const float bv = bias[o_base + m];
        #pragma unroll
        for (int i = 0; i < 8; ++i) {           // C/D: row = half*8 + i, col = m
            float v = c[i] + bv;
            if (doRelu) v = fmaxf(v, 0.0f);
            out[(long long)(node_base + half * 8 + i) * D_FEAT + o_base + m] = v;
        }
    }
}

// ---------------------------------------------------------------------------
// Edge predictor: 16 pairs/block. cat(256) ->WMMA P1(relu)->128 ->WMMA P2(relu)
// ->128 -> dot with P3 (8 lanes/pair + wave shuffle reduce).
// ---------------------------------------------------------------------------
__global__ __launch_bounds__(128)
void predict_kernel(const float* __restrict__ h,
                    const int* __restrict__ srcI, const int* __restrict__ dstI,
                    const __bf16* __restrict__ P1w, const float* __restrict__ P1b,
                    const __bf16* __restrict__ P2w, const float* __restrict__ P2b,
                    const float* __restrict__ P3w, const float* __restrict__ P3b,
                    float* __restrict__ out)
{
    __shared__ __align__(16) __bf16 feat[16 * 256];
    __shared__ __align__(16) __bf16 hid [16 * 128];
    __shared__ __align__(16) __bf16 hid2[16 * 128];
    const int tid = threadIdx.x;
    const int pair_base = blockIdx.x * 16;

    // Gather cat(h[src], h[dst]) as bf16: 16 rows x 256 ch, 4 per iteration.
    for (int i = tid; i < 16 * 64; i += 128) {
        int r = i >> 6, q = (i & 63) * 4;
        int node = (q < 128) ? srcI[pair_base + r] : dstI[pair_base + r];
        float4 v = *(const float4*)(h + (long long)node * D_FEAT + (q & 127));
        store_bf4(&feat[r * 256 + q], v);
    }
    __syncthreads();

    const int lane = tid & 31, wave = tid >> 5;
    const int m = lane & 15, half = lane >> 4;

    // P1: 256 -> 128, relu
    #pragma unroll
    for (int t = 0; t < 2; ++t) {
        const int o_base = (wave + t * 4) * 16;
        v8f c = {};
        #pragma unroll
        for (int kb = 0; kb < 8; ++kb) {
            v16bf a = load_a_frag(feat, 256, m, half, kb);
            v16bf b = load_b_frag(P1w, 256, o_base + m, half, kb);
            c = __builtin_amdgcn_wmma_f32_16x16x32_bf16(false, a, false, b,
                                                        (short)0, c, false, false);
        }
        const float bv = P1b[o_base + m];
        #pragma unroll
        for (int i = 0; i < 8; ++i)
            hid[(half * 8 + i) * 128 + o_base + m] = (__bf16)fmaxf(c[i] + bv, 0.0f);
    }
    __syncthreads();

    // P2: 128 -> 128, relu
    #pragma unroll
    for (int t = 0; t < 2; ++t) {
        const int o_base = (wave + t * 4) * 16;
        v8f c = {};
        #pragma unroll
        for (int kb = 0; kb < 4; ++kb) {
            v16bf a = load_a_frag(hid, 128, m, half, kb);
            v16bf b = load_b_frag(P2w, 128, o_base + m, half, kb);
            c = __builtin_amdgcn_wmma_f32_16x16x32_bf16(false, a, false, b,
                                                        (short)0, c, false, false);
        }
        const float bv = P2b[o_base + m];
        #pragma unroll
        for (int i = 0; i < 8; ++i)
            hid2[(half * 8 + i) * 128 + o_base + m] = (__bf16)fmaxf(c[i] + bv, 0.0f);
    }
    __syncthreads();

    // P3: 128 -> 1. 8 lanes per pair, each sums 16, then shuffle-reduce in wave.
    {
        int p = tid >> 3, seg = tid & 7;
        const __bf16* hp = hid2 + p * 128 + seg * 16;
        const float*  wp = P3w + seg * 16;
        float acc = 0.0f;
        #pragma unroll
        for (int k = 0; k < 16; ++k) acc += (float)hp[k] * wp[k];
        #pragma unroll
        for (int off = 4; off > 0; off >>= 1) acc += __shfl_xor(acc, off, 8);
        if (seg == 0) out[pair_base + p] = acc + P3b[0];
    }
}

// ---------------------------------------------------------------------------
extern "C" void kernel_launch(void* const* d_in, const int* in_sizes, int n_in,
                              void* d_out, int out_size, void* d_ws, size_t ws_size,
                              hipStream_t stream)
{
    const float* x    = (const float*)d_in[0];
    const int*   esrc = (const int*)d_in[1];
    const int*   edst = (const int*)d_in[2];
    const int*   psrc = (const int*)d_in[3];
    const int*   pdst = (const int*)d_in[4];
    const int*   nsrc = (const int*)d_in[5];
    const int*   ndst = (const int*)d_in[6];
    const float* W1n  = (const float*)d_in[7];
    const float* W1s  = (const float*)d_in[8];
    const float* b1   = (const float*)d_in[9];
    const float* W2n  = (const float*)d_in[10];
    const float* W2s  = (const float*)d_in[11];
    const float* b2   = (const float*)d_in[12];
    const float* P1w  = (const float*)d_in[13];
    const float* P1b  = (const float*)d_in[14];
    const float* P2w  = (const float*)d_in[15];
    const float* P2b  = (const float*)d_in[16];
    const float* P3w  = (const float*)d_in[17];
    const float* P3b  = (const float*)d_in[18];
    float* outp = (float*)d_out;

    const int nN = in_sizes[0] / D_FEAT;   // 100000 (divisible by 16)
    const int nE = in_sizes[1];            // 1600000
    const int nP = in_sizes[3];            // 100000 (divisible by 16)

    // Workspace carve-out (~154 MB + ~229 KB bf16 weights; L2 is 192 MB).
    char* ws = (char*)d_ws;
    float* msg  = (float*)ws;                                   // nN*128 f32
    float* degv = (float*)(ws + (size_t)nN * D_FEAT * 4);       // nN f32
    float* h1   = (float*)(ws + (size_t)nN * D_FEAT * 4 + (size_t)nN * 4);
    float* h2   = h1 + (size_t)nN * D_FEAT;
    __bf16* W1s_b = (__bf16*)(h2 + (size_t)nN * D_FEAT);
    __bf16* W1n_b = W1s_b + 128 * 128;
    __bf16* W2s_b = W1n_b + 128 * 128;
    __bf16* W2n_b = W2s_b + 128 * 128;
    __bf16* P1w_b = W2n_b + 128 * 128;
    __bf16* P2w_b = P1w_b + 128 * 256;

    const long long featElems = (long long)nN * D_FEAT;
    dim3 blk256(256), blk128(128);
    dim3 gZeroF((unsigned)((featElems + 255) / 256));
    dim3 gZeroD((unsigned)((nN + 255) / 256));
    dim3 gDeg((unsigned)((nE + 255) / 256));
    dim3 gScat((unsigned)(((long long)nE * 32 + 255) / 256));
    dim3 gLayer((unsigned)(nN / 16));
    dim3 gPred((unsigned)(nP / 16));
    dim3 gCvt16k((16384 / 4 + 255) / 256);
    dim3 gCvt32k((32768 / 4 + 255) / 256);

    // Weight conversion f32 -> bf16 (tiny, once per launch)
    f32_to_bf16_kernel<<<gCvt16k, blk256, 0, stream>>>(W1s, W1s_b, 16384);
    f32_to_bf16_kernel<<<gCvt16k, blk256, 0, stream>>>(W1n, W1n_b, 16384);
    f32_to_bf16_kernel<<<gCvt16k, blk256, 0, stream>>>(W2s, W2s_b, 16384);
    f32_to_bf16_kernel<<<gCvt16k, blk256, 0, stream>>>(W2n, W2n_b, 16384);
    f32_to_bf16_kernel<<<gCvt32k, blk256, 0, stream>>>(P1w, P1w_b, 32768);
    f32_to_bf16_kernel<<<gCvt16k, blk256, 0, stream>>>(P2w, P2w_b, 16384);

    // Layer 1
    zero_f32_kernel<<<gZeroF, blk256, 0, stream>>>(msg, featElems);
    zero_f32_kernel<<<gZeroD, blk256, 0, stream>>>(degv, nN);
    deg_kernel<<<gDeg, blk256, 0, stream>>>(edst, degv, nE);
    scatter_kernel<<<gScat, blk256, 0, stream>>>(x, esrc, edst, msg, nE);
    sage_layer_kernel<<<gLayer, blk128, 0, stream>>>(x, msg, degv, W1s_b, W1n_b, b1, h1, 1);

    // Layer 2 (deg reused — same graph)
    zero_f32_kernel<<<gZeroF, blk256, 0, stream>>>(msg, featElems);
    scatter_kernel<<<gScat, blk256, 0, stream>>>(h1, esrc, edst, msg, nE);
    sage_layer_kernel<<<gLayer, blk128, 0, stream>>>(h1, msg, degv, W2s_b, W2n_b, b2, h2, 0);

    // Predictor: pos then neg (outputs concatenated)
    predict_kernel<<<gPred, blk128, 0, stream>>>(h2, psrc, pdst, P1w_b, P1b, P2w_b, P2b,
                                                 P3w, P3b, outp);
    predict_kernel<<<gPred, blk128, 0, stream>>>(h2, nsrc, ndst, P1w_b, P1b, P2w_b, P2b,
                                                 P3w, P3b, outp + nP);
}
